// NeuralCDEModel_20023137534271
// MI455X (gfx1250) — matrix operand order repaired and checked by reading, not compiled, not run on hardware
//
#include <hip/hip_runtime.h>
#include <hip/hip_bf16.h>

// Problem constants (match reference)
#define B_TOT 1024
#define T_N   11
#define C_N   64
#define H_N   256
#define N2    (H_N * C_N)     // 16384 columns of W2
#define NTHREADS 512          // 16 waves of 32
#define KB_N  (H_N / 32)      // 8 K-blocks of 32 along the contraction dim

typedef __attribute__((ext_vector_type(16))) __bf16 v16bf;
typedef __attribute__((ext_vector_type(8)))  float  v8f;

// ---------------------------------------------------------------------------
// Operand-layout helpers (ISA 7.12.2, 16-bit WMMA, wave32):
//  A (16x32): lane holds row M=lane%16; vgpr e half h -> K = (2e+h) for e<4,
//             (2e+8+h) for e>=4, +8 for lanes 16..31.  i=2e+h => off = (i<8?i:i+8)+8*hi
//  B (32x16): lane holds col N=lane%16; element i holds K = (i<8?i:i+8) + 8*(lane>=16)
//             -> a swizzled B tile is 16 bf16 per lane (32B), same K order as A.
// ---------------------------------------------------------------------------
__device__ __forceinline__ v16bf build_a_from_bf16(const __bf16* __restrict__ row, int hi) {
    v16bf a;
    #pragma unroll
    for (int i = 0; i < 16; ++i) {
        const int off = ((i < 8) ? i : i + 8) + 8 * hi;
        a[i] = row[off];
    }
    return a;
}
__device__ __forceinline__ v16bf build_a_from_f32(const float* __restrict__ row, int hi) {
    v16bf a;
    #pragma unroll
    for (int i = 0; i < 16; ++i) {
        const int off = ((i < 8) ? i : i + 8) + 8 * hi;
        a[i] = (__bf16)row[off];
    }
    return a;
}

// ---------------------------------------------------------------------------
// One-time swizzle: src (K x N, f32 row-major) -> bf16 WMMA-B tiles.
// dst[((ct*(K/32)+kb)*32 + lane)*16 + i] = src[(kb*32 + (i<8?i:i+8) + 8*hi)*N + ct*16 + n]
// ---------------------------------------------------------------------------
__global__ void swizzle_bf16_kernel(const float* __restrict__ src,
                                    __bf16* __restrict__ dst, int K, int N) {
    const int g    = blockIdx.x * blockDim.x + threadIdx.x;
    const int lane = g & 31;
    const int tile = g >> 5;
    const int kbn  = K >> 5;
    const int ct   = tile / kbn;
    const int kb   = tile % kbn;
    if (ct >= (N >> 4)) return;
    const int n  = lane & 15;
    const int hi = lane >> 4;
    const int col = ct * 16 + n;
    __bf16* o = dst + ((size_t)tile * 32 + lane) * 16;
    #pragma unroll
    for (int i = 0; i < 16; ++i) {
        const int k = kb * 32 + ((i < 8) ? i : i + 8) + 8 * hi;
        o[i] = (__bf16)src[(size_t)k * N + col];
    }
}

// ---------------------------------------------------------------------------
// Main kernel: each workgroup owns a 16-row batch tile and integrates all
// 10 RK4 steps locally.  16 waves; wave w owns hidden cols / j-tile range.
// ---------------------------------------------------------------------------
__global__ __launch_bounds__(NTHREADS, 1)
void ncde_fused_kernel(const float* __restrict__ coeffs,  // (B, T-1, 4C)
                       const float* __restrict__ ts,      // (T,)
                       const float* __restrict__ W_init,  // (C, H)
                       const float* __restrict__ b_init,  // (H,)
                       const float* __restrict__ b1,      // (H,)
                       const float* __restrict__ b2,      // (H*C,)
                       const float* __restrict__ W_out,   // (H, C)
                       const float* __restrict__ b_out,   // (C,)
                       const __bf16* __restrict__ w1sw,   // swizzled W1 (H x H)
                       const __bf16* __restrict__ w2sw,   // swizzled W2 (H x H*C)
                       float* __restrict__ out)           // (B, T, C)
{
    __shared__ float  z_s  [16][H_N + 1];    // stride 257 -> conflict-free column reads
    __shared__ float  zin_s[16][H_N + 1];
    __shared__ float  acc_s[16][H_N + 1];
    __shared__ __bf16 h_s  [16][H_N + 2];
    __shared__ float  dX_s [16][C_N + 4];

    const int tid    = threadIdx.x;
    const int lane   = tid & 31;
    const int wave   = tid >> 5;
    const int laneHi = lane >> 4;
    const int mA     = lane & 15;
    const int nB     = lane & 15;
    const int b0     = blockIdx.x * 16;

    float tg[T_N];
    #pragma unroll
    for (int q = 0; q < T_N; ++q) tg[q] = ts[q];

    // ---------------- init: z0 = a[:,0] @ W_init + b_init ----------------
    #pragma unroll 1
    for (int r = 0; r < 8; ++r) {
        int o = tid + r * NTHREADS;
        int m = o >> 8, n = o & 255;
        const float* cf = coeffs + (size_t)(b0 + m) * (T_N - 1) * (4 * C_N);
        float s = b_init[n];
        #pragma unroll 4
        for (int c = 0; c < C_N; ++c) s += cf[c] * W_init[c * H_N + n];
        z_s[m][n]   = s;
        zin_s[m][n] = s;
    }
    __syncthreads();

    // out row t=0
    #pragma unroll 1
    for (int r = 0; r < 2; ++r) {
        int o = tid + r * NTHREADS;
        int m = o >> 6, c = o & 63;
        float s = b_out[c];
        #pragma unroll 4
        for (int n = 0; n < H_N; ++n) s += z_s[m][n] * W_out[n * C_N + c];
        out[((size_t)(b0 + m) * T_N + 0) * C_N + c] = s;
    }
    __syncthreads();

    const float b1v = b1[wave * 16 + nB];

    // ---------------- time integration ----------------
    #pragma unroll 1
    for (int i = 0; i < T_N - 1; ++i) {
        const float t0 = tg[i];
        const float dt = tg[i + 1] - t0;

        #pragma unroll 1
        for (int st = 0; st < 4; ++st) {
            const float tev = (st == 0) ? t0 : ((st == 3) ? (t0 + dt) : (t0 + 0.5f * dt));

            // bucket (searchsorted-right - 1, clipped)
            int   idx  = 0;
            float tsel = tg[0];
            #pragma unroll
            for (int q = 1; q <= T_N - 2; ++q) {
                bool le = (tg[q] <= tev);
                idx  = le ? q     : idx;
                tsel = le ? tg[q] : tsel;
            }
            const float frac = tev - tsel;

            // ---- spline derivative dX (16 x 64) ----
            #pragma unroll 1
            for (int r = 0; r < 2; ++r) {
                int o = tid + r * NTHREADS;
                int m = o >> 6, c = o & 63;
                const float* cf = coeffs + (((size_t)(b0 + m) * (T_N - 1)) + idx) * (4 * C_N);
                dX_s[m][c] = cf[C_N + c] + (cf[2 * C_N + c] + cf[3 * C_N + c] * frac) * frac;
            }

            // ---- h = tanh(zin @ W1 + b1) via WMMA; wave w -> hidden cols [16w,16w+16) ----
            {
                v16bf azin[KB_N];
                #pragma unroll
                for (int kb = 0; kb < KB_N; ++kb)
                    azin[kb] = build_a_from_f32(&zin_s[mA][kb * 32], laneHi);
                v8f hacc = {};
                #pragma unroll
                for (int kb = 0; kb < KB_N; ++kb) {
                    const v16bf bv = *(const v16bf*)(w1sw +
                        ((size_t)(wave * KB_N + kb) * 32 + lane) * 16);
                    hacc = __builtin_amdgcn_wmma_f32_16x16x32_bf16(
                        false, azin[kb], false, bv, (short)0, hacc, false, false);
                }
                #pragma unroll
                for (int r = 0; r < 8; ++r) {
                    const int M = r + 8 * laneHi;
                    h_s[M][wave * 16 + nB] = (__bf16)tanhf(hacc[r] + b1v);
                }
            }
            __syncthreads();   // h_s + dX_s ready; zin_s consumption finished

            // ---- g = einsum(f, dX):  F = h @ W2 tile-wise, contracted on the fly ----
            // A operands (h k-blocks): built ONCE per eval, reused for 64 column tiles.
            v16bf ha[KB_N];
            #pragma unroll
            for (int kb = 0; kb < KB_N; ++kb)
                ha[kb] = build_a_from_bf16(&h_s[mA][kb * 32], laneHi);

            #pragma unroll 1
            for (int jj = 0; jj < 16; ++jj) {
                const int j = wave * 16 + jj;

                // kb-outer / cb-inner, fully unrolled: 32 WMMAs, 64 b128 loads in
                // one scheduling window, 4 independent accumulators, shared A.
                v8f acc[4];
                #pragma unroll
                for (int cb = 0; cb < 4; ++cb) acc[cb] = (v8f){};
                #pragma unroll
                for (int kb = 0; kb < KB_N; ++kb) {
                    #pragma unroll
                    for (int cb = 0; cb < 4; ++cb) {
                        const v16bf bv = *(const v16bf*)(w2sw +
                            (((size_t)(j * 4 + cb) * KB_N + kb) * 32 + lane) * 16);
                        acc[cb] = __builtin_amdgcn_wmma_f32_16x16x32_bf16(
                            false, ha[kb], false, bv, (short)0, acc[cb], false, false);
                    }
                }

                // epilogue: partial[r] = sum_c (F + b2) * dX over this lane's columns
                float partial[8] = {};
                #pragma unroll
                for (int cb = 0; cb < 4; ++cb) {
                    const int   cc  = cb * 16 + nB;
                    const float b2v = b2[(size_t)j * C_N + cc];
                    #pragma unroll
                    for (int r = 0; r < 8; ++r)
                        partial[r] += (acc[cb][r] + b2v) * dX_s[r + 8 * laneHi][cc];
                }
                // reduce over the 16 columns (stays within each 16-lane half)
                #pragma unroll
                for (int r = 0; r < 8; ++r) {
                    float p = partial[r];
                    p += __shfl_xor(p, 1, 32);
                    p += __shfl_xor(p, 2, 32);
                    p += __shfl_xor(p, 4, 32);
                    p += __shfl_xor(p, 8, 32);
                    partial[r] = p;
                }
                if (nB == 0) {   // lanes 0 (rows 0-7) and 16 (rows 8-15) commit column j
                    #pragma unroll
                    for (int r = 0; r < 8; ++r) {
                        const int   M = r + 8 * laneHi;
                        const float g = partial[r];
                        if (st == 0) {
                            zin_s[M][j] = z_s[M][j] + 0.5f * dt * g;
                            acc_s[M][j] = g;
                        } else if (st == 1) {
                            zin_s[M][j] = z_s[M][j] + 0.5f * dt * g;
                            acc_s[M][j] += 2.0f * g;
                        } else if (st == 2) {
                            zin_s[M][j] = z_s[M][j] + dt * g;
                            acc_s[M][j] += 2.0f * g;
                        } else {
                            const float zn = z_s[M][j] +
                                (dt * (1.0f / 6.0f)) * (acc_s[M][j] + g);
                            z_s[M][j]   = zn;
                            zin_s[M][j] = zn;
                        }
                    }
                }
            }
            __syncthreads();
        } // stage

        // ---- out row t=i+1 : z @ W_out + b_out ----
        #pragma unroll 1
        for (int r = 0; r < 2; ++r) {
            int o = tid + r * NTHREADS;
            int m = o >> 6, c = o & 63;
            float s = b_out[c];
            #pragma unroll 4
            for (int n = 0; n < H_N; ++n) s += z_s[m][n] * W_out[n * C_N + c];
            out[((size_t)(b0 + m) * T_N + (i + 1)) * C_N + c] = s;
        }
        __syncthreads();
    } // time step
}

extern "C" void kernel_launch(void* const* d_in, const int* in_sizes, int n_in,
                              void* d_out, int out_size, void* d_ws, size_t ws_size,
                              hipStream_t stream) {
    (void)in_sizes; (void)n_in; (void)out_size; (void)ws_size;
    const float* coeffs = (const float*)d_in[0];
    const float* ts     = (const float*)d_in[1];
    const float* W_init = (const float*)d_in[2];
    const float* b_init = (const float*)d_in[3];
    const float* W1     = (const float*)d_in[4];
    const float* b1     = (const float*)d_in[5];
    const float* W2     = (const float*)d_in[6];
    const float* b2     = (const float*)d_in[7];
    const float* W_out  = (const float*)d_in[8];
    const float* b_out  = (const float*)d_in[9];
    float* out = (float*)d_out;

    // Workspace: bf16 swizzled W1 (128 tiles) then W2 (8192 tiles), 1 KB each (~8.2 MB).
    __bf16* w1sw = (__bf16*)d_ws;
    __bf16* w2sw = w1sw + (size_t)(H_N / 16) * (H_N / 32) * 512;   // 128 * 512 halves

    {   // W1: K=256, N=256 -> 128 tiles * 32 lanes = 4096 threads
        dim3 g((H_N / 16) * (H_N / 32) * 32 / 256), b(256);
        swizzle_bf16_kernel<<<g, b, 0, stream>>>(W1, w1sw, H_N, H_N);
    }
    {   // W2: K=256, N=16384 -> 8192 tiles * 32 lanes = 262144 threads
        dim3 g((N2 / 16) * (H_N / 32) * 32 / 256), b(256);
        swizzle_bf16_kernel<<<g, b, 0, stream>>>(W2, w2sw, H_N, N2);
    }

    dim3 grid(B_TOT / 16), block(NTHREADS);
    ncde_fused_kernel<<<grid, block, 0, stream>>>(coeffs, ts, W_init, b_init,
                                                  b1, b2, W_out, b_out,
                                                  w1sw, w2sw, out);
}
